// PUConLoss_13606456393936
// MI455X (gfx1250) — matrix operand order repaired and checked by reading, not compiled, not run on hardware
//
#include <hip/hip_runtime.h>
#include <hip/hip_bf16.h>
#include <math.h>

// ---------------------------------------------------------------------------
// PUCon loss, fused for MI455X (gfx1250, wave32, WMMA f32_16x16x32_f16).
//   B = 2048, D = 128, 2B = 4096 rows, TEMP = 0.5 (scale = 2).
//
// Pipeline:
//   init      : zero per-row partials, count positive labels
//   normalize : row-normalize [z; z_aug] -> f16 (1 MB, L2-resident)
//   main      : 256 blocks = 32 row-blocks (128 rows, 8 waves x 16-row strip)
//               x 8 column chunks (512 cols). B tiles double-buffered in LDS,
//               shared by all 8 waves; per-row partial exp-sums atomically
//               accumulated into workspace (sums are additive across chunks).
//   finalize  : per-row log + sup/unsup combine -> scalar loss
//
// ws layout: [0..3]      int cnt1
//            [64..)      float rowAcc[3*4096]  (sExp | sEq | pos)
//            [65536..)   Zn16: 4096 x 128 _Float16
// ---------------------------------------------------------------------------

typedef _Float16 v16h __attribute__((ext_vector_type(16)));
typedef _Float16 v8h  __attribute__((ext_vector_type(8)));
typedef _Float16 v4h  __attribute__((ext_vector_type(4)));
typedef float    v8f  __attribute__((ext_vector_type(8)));

#define PB     2048
#define PD     128
#define PN2    4096
#define CCHUNK 512           // columns per block (32 tiles of 16)
#define LDSPAD 136           // 128 halves + 8 pad -> 272B row stride, no bank conflicts

// ---- init: zero per-row partials, count positive labels --------------------
__global__ void pucon_init(const int* __restrict__ labels,
                           int* __restrict__ cntp, float* __restrict__ rowAcc) {
  __shared__ int sh[256];
  int t = threadIdx.x;
  for (int i = t; i < 3 * PN2; i += 256) rowAcc[i] = 0.0f;
  int s = 0;
  for (int i = t; i < PB; i += 256) s += labels[i];
  sh[t] = s;
  __syncthreads();
  for (int stride = 128; stride > 0; stride >>= 1) {
    if (t < stride) sh[t] += sh[t + stride];
    __syncthreads();
  }
  if (t == 0) cntp[0] = sh[0];
}

// ---- normalize rows of [z; z_aug] into f16 ---------------------------------
__global__ __launch_bounds__(256) void pucon_normalize(
    const float* __restrict__ z, const float* __restrict__ za,
    _Float16* __restrict__ zn) {
  int wave = threadIdx.x >> 5;
  int lane = threadIdx.x & 31;
  int row  = blockIdx.x * 8 + wave;                 // 512 blocks -> 4096 rows
  const float* src = (row < PB) ? (z + (size_t)row * PD)
                                : (za + (size_t)(row - PB) * PD);
  float4 x = ((const float4*)src)[lane];
  float ss = x.x * x.x + x.y * x.y + x.z * x.z + x.w * x.w;
  for (int off = 16; off > 0; off >>= 1) ss += __shfl_xor(ss, off);
  float inv = 1.0f / fmaxf(sqrtf(ss), 1e-12f);
  v4h o;
  o[0] = (_Float16)(x.x * inv);
  o[1] = (_Float16)(x.y * inv);
  o[2] = (_Float16)(x.z * inv);
  o[3] = (_Float16)(x.w * inv);
  *(v4h*)(zn + (size_t)row * PD + lane * 4) = o;
}

// ---- fused similarity + partial row reductions -----------------------------
__global__ __launch_bounds__(256) void pucon_main(
    const _Float16* __restrict__ zn, const int* __restrict__ labels,
    float* __restrict__ rowAcc) {
  __shared__ __align__(16) _Float16 bt[2][16][LDSPAD];  // double-buffered B tile

  const int tid  = threadIdx.x;
  const int lane = tid & 31;
  const int wv   = tid >> 5;                  // 8 waves per block
  const int rowBlk = blockIdx.x & 31;         // 32 row-blocks
  const int colBlk = blockIdx.x >> 5;         // 8 column chunks
  const int i0      = rowBlk * 128 + wv * 16; // this wave's 16-row strip
  const int colBase = colBlk * CCHUNK;

  const bool hi = lane >= 16;
  const int rowBase = i0 + (hi ? 8 : 0);      // C-fragment row group
  const int nCol    = lane & 15;              // C-fragment column

  // staging role: thread moves 8 halves (16B) of the 16x128 tile
  const int sr = tid >> 4;                    // tile row 0..15
  const int sc = tid & 15;                    // 16B chunk 0..15

  // A fragments (16x32 f16 per K-chunk, K=128 -> 4), loaded once from global.
  v16h a[4];
  {
    const _Float16* ap = zn + (size_t)(i0 + (lane & 15)) * PD + (hi ? 8 : 0);
#pragma unroll
    for (int k = 0; k < 4; ++k) {
      v8h lo = *(const v8h*)(ap + 32 * k);
      v8h hh = *(const v8h*)(ap + 32 * k + 16);
      union { v16h v; v8h h[2]; } u;
      u.h[0] = lo;
      u.h[1] = hh;
      a[k] = u.v;
    }
  }

  int   labRow[8];
  float sExp[8], sEq[8], posV[8];
#pragma unroll
  for (int r = 0; r < 8; ++r) {
    labRow[r] = labels[(rowBase + r) & (PB - 1)];
    sExp[r] = 0.0f;
    sEq[r]  = 0.0f;
    posV[r] = 0.0f;
  }

  // stage tile 0 of this block's column chunk
  *(v8h*)(&bt[0][sr][sc * 8]) =
      *(const v8h*)(zn + (size_t)(colBase + sr) * PD + sc * 8);
  __syncthreads();

  const int NT = CCHUNK / 16;                 // 32 column tiles
  for (int jt = 0; jt < NT; ++jt) {
    const int p = jt & 1;
    const int j = colBase + (jt << 4);

    // issue next tile's global loads early (overlap with WMMA + exp work)
    v8h nxt;
    const bool have = (jt + 1) < NT;
    if (have)
      nxt = *(const v8h*)(zn + (size_t)(j + 16 + sr) * PD + sc * 8);

    // B fragments from LDS: lane<16 -> col j+lane, K {kb..kb+7, kb+16..kb+23}; hi -> +8
    const _Float16* bp = &bt[p][nCol][hi ? 8 : 0];
    v8f cacc = {0.0f, 0.0f, 0.0f, 0.0f, 0.0f, 0.0f, 0.0f, 0.0f};
#pragma unroll
    for (int k = 0; k < 4; ++k) {
      v8h lo = *(const v8h*)(bp + 32 * k);
      v8h hh = *(const v8h*)(bp + 32 * k + 16);
      union { v16h v; v8h h[2]; } u;
      u.h[0] = lo;
      u.h[1] = hh;
      cacc = __builtin_amdgcn_wmma_f32_16x16x32_f16(
          false, a[k], false, u.v, (short)0, cacc, false, false);
    }

    const int col  = j + nCol;
    const int labC = labels[col & (PB - 1)];
#pragma unroll
    for (int r = 0; r < 8; ++r) {
      const float v   = cacc[r] * 2.0f;       // / TEMP
      const int   row = rowBase + r;
      const bool  nd  = (col != row);
      // fixed shift 2.0 == exact row max (cos/temp <= 2); LSE shift-invariant
      const float e = __expf(v - 2.0f);
      sExp[r] += nd ? e : 0.0f;
      sEq[r]  += (nd && (labC == labRow[r])) ? v : 0.0f;
      posV[r] += (col == ((row + PB) & (PN2 - 1))) ? v : 0.0f;
    }

    __syncthreads();                          // all reads of bt[p] done
    if (have) *(v8h*)(&bt[p ^ 1][sr][sc * 8]) = nxt;
    __syncthreads();                          // stores visible
  }

  // reduce across the 16 lanes sharing each row group
  for (int off = 1; off < 16; off <<= 1) {
#pragma unroll
    for (int r = 0; r < 8; ++r) {
      sExp[r] += __shfl_xor(sExp[r], off);
      sEq[r]  += __shfl_xor(sEq[r], off);
      posV[r] += __shfl_xor(posV[r], off);
    }
  }

  // lanes 0 and 16 publish this chunk's partials for their 8 rows
  if ((lane & 15) == 0) {
    float* sExpG = rowAcc;
    float* sEqG  = rowAcc + PN2;
    float* posG  = rowAcc + 2 * PN2;
#pragma unroll
    for (int r = 0; r < 8; ++r) {
      const int row = rowBase + r;
      atomicAdd(&sExpG[row], sExp[r]);
      atomicAdd(&sEqG[row], sEq[r]);
      atomicAdd(&posG[row], posV[r]);
    }
  }
}

// ---- finalize: per-row log + combine ---------------------------------------
__global__ void pucon_finalize(const float* __restrict__ rowAcc,
                               const int* __restrict__ labels,
                               const int* __restrict__ cntp,
                               float* __restrict__ out) {
  __shared__ float shm[256], shu[256];
  const int t = threadIdx.x;
  const float* sExpG = rowAcc;
  const float* sEqG  = rowAcc + PN2;
  const float* posG  = rowAcc + 2 * PN2;
  const int cnt1 = cntp[0];
  float cm = 0.0f, cu = 0.0f;
  for (int r = t; r < PN2; r += 256) {
    const int   lab   = labels[r & (PB - 1)];
    const float lse   = __logf(sExpG[r]) + 2.0f;
    const float unsup = lse - posG[r];
    const float cntEqOff = (float)(2 * (lab ? cnt1 : (PB - cnt1)) - 1);
    const float sup   = lse - sEqG[r] / cntEqOff;
    cm += lab ? sup : unsup;
    cu += unsup;
  }
  shm[t] = cm;
  shu[t] = cu;
  __syncthreads();
  for (int s = 128; s > 0; s >>= 1) {
    if (t < s) {
      shm[t] += shm[t + s];
      shu[t] += shu[t + s];
    }
    __syncthreads();
  }
  if (t == 0) out[0] = ((cnt1 == 0) ? shu[0] : shm[0]) / (float)PN2;
}

// ---------------------------------------------------------------------------
extern "C" void kernel_launch(void* const* d_in, const int* in_sizes, int n_in,
                              void* d_out, int out_size, void* d_ws, size_t ws_size,
                              hipStream_t stream) {
  const float* z      = (const float*)d_in[0];
  const float* za     = (const float*)d_in[1];
  const int*   labels = (const int*)d_in[2];

  int*      cntp   = (int*)d_ws;
  float*    rowAcc = (float*)((char*)d_ws + 64);       // 3*4096 floats
  _Float16* zn     = (_Float16*)((char*)d_ws + 65536); // 1 MB

  pucon_init<<<1, 256, 0, stream>>>(labels, cntp, rowAcc);
  pucon_normalize<<<PN2 / 8, 256, 0, stream>>>(z, za, zn);
  pucon_main<<<(PN2 / 128) * (PN2 / CCHUNK), 256, 0, stream>>>(zn, labels, rowAcc);
  pucon_finalize<<<1, 256, 0, stream>>>(rowAcc, labels, cntp, (float*)d_out);
}
